// SeqLSTM_78202764526011
// MI455X (gfx1250) — matrix operand order repaired and checked by reading, not compiled, not run on hardware
//
#include <hip/hip_runtime.h>
#include <hip/hip_bf16.h>

// ---------------------------------------------------------------------------
// Problem constants (from reference): IN_DIM=64, HID=512, B=256, T=512
// ---------------------------------------------------------------------------
#define IN_DIM 64
#define HID    512
#define BATCH  256
#define SEQ_T  512
#define GATES  (4 * HID)        // 2048
#define BT     16               // batch rows per workgroup (== WMMA M)
#define NW     16               // waves per workgroup
#define NTHR   (NW * 32)        // 512 threads
#define KCAT   (IN_DIM + HID)   // 576: layer-0 concatenated K = [x | h0]
#define LDA0   584              // LDS stride of A0 = [x|h0] rows (mult of 8)
#define LDH1   520              // LDS stride of h1 rows (mult of 8)

// workspace layout (bf16 element offsets, then fp32 bias region)
#define OFS_WC0  0
#define N_WC0    (GATES * KCAT)            // 1179648  [W_ih0 | W_hh0] packed
#define OFS_WIH1 (OFS_WC0 + N_WC0)
#define N_WHH    (GATES * HID)             // 1048576
#define OFS_WHH1 (OFS_WIH1 + N_WHH)
#define N_BF_TOT (OFS_WHH1 + N_WHH)        // 3276800 bf16 elems
#define OFS_BIAS0_BYTES ((size_t)N_BF_TOT * 2)
#define OFS_BIAS1_BYTES (OFS_BIAS0_BYTES + (size_t)GATES * 4)

// LDS layout (bytes); all offsets 16B aligned
#define SM_A0A   0
#define SM_A0B   (SM_A0A + BT * LDA0 * 2)   // +18688
#define SM_H1A   (SM_A0B + BT * LDA0 * 2)   // 37376
#define SM_H1B   (SM_H1A + BT * LDH1 * 2)   // +16640
#define SM_C0    (SM_H1B + BT * LDH1 * 2)   // 70656
#define SM_C1    (SM_C0 + BT * HID * 4)     // +32768
#define SM_TOTAL (SM_C1 + BT * HID * 4)     // 136192 bytes

typedef __attribute__((ext_vector_type(16))) __bf16 v16bf;
typedef __attribute__((ext_vector_type(8)))  __bf16 v8bf;
typedef __attribute__((ext_vector_type(8)))  float  v8f;

#define LOG2E 1.4426950408889634f

// sigmoid via hardware transcendentals: v_exp_f32 + v_rcp_f32 (no IEEE divide)
__device__ __forceinline__ float fast_sigmoid(float x) {
    return __builtin_amdgcn_rcpf(1.0f + __builtin_amdgcn_exp2f(-LOG2E * x));
}
__device__ __forceinline__ float fast_tanh(float x) {
    return __builtin_fmaf(2.0f, fast_sigmoid(2.0f * x), -1.0f);
}
__device__ __forceinline__ v8bf ld8(const __bf16* p) {
    return *(const v8bf*)p;
}
__device__ __forceinline__ v16bf make_frag(v8bf lo, v8bf hi) {
    return __builtin_shufflevector(lo, hi, 0,1,2,3,4,5,6,7,8,9,10,11,12,13,14,15);
}
__device__ __forceinline__ v8f splat8(float v) {
    v8f r;
    #pragma unroll
    for (int i = 0; i < 8; ++i) r[i] = v;
    return r;
}

#define WMMA_BF16(A, B, C) \
    __builtin_amdgcn_wmma_f32_16x16x32_bf16(false, (A), false, (B), (short)0, (C), false, false)

// ---------------------------------------------------------------------------
// One LSTM cell step for a 16(batch) x 16(hidden-col) tile, all four gates,
// as up to two GEMM segments:   gates = bias + A1(16xK1)@W1^T + A2(16xK2)@W2^T
// A-frag (16x32 bf16): lane holds row M=lane%16; 16 contiguous K at ka=(lane<16?0:16)
// B-frag (32x16 bf16): lane holds col N=lane%16; 16 contiguous K at kb=(lane<16?0:16)
// C/D frag: VGPR r, lane L -> (M = r + (L<16?0:8), N = L%16)
// Weights indexed off the uniform base with 32-bit element offsets ->
// global_load_b128; A from LDS -> ds_load_b128. All loops have loop-varying
// addresses so nothing gets LICM-hoisted/spilled.
// ---------------------------------------------------------------------------
template<int K1, int LD_A1, int LDW1, int K2, int LD_A2, int LDW2, int LDO>
__device__ __forceinline__ void lstm_tile_step(
    const __bf16* a1,                     // [BT][LD_A1] LDS (segment 1)
    const __bf16* __restrict__ w1,        // [GATES][LDW1] bf16 global
    const __bf16* a2,                     // [BT][LD_A2] LDS (segment 2, opt)
    const __bf16* __restrict__ w2,        // [GATES][LDW2] bf16 global (opt)
    const float* __restrict__ bias,       // [GATES] fp32 global
    float* cst,                           // [BT][HID] fp32 LDS cell state
    __bf16* hout,                         // [BT][LDO] LDS h output
    int jbase)
{
    const int  lane = threadIdx.x & 31;
    const int  n    = lane & 15;
    const bool lo16 = lane < 16;
    const int  ka   = lo16 ? 0 : 16;      // A-fragment K chunk base
    const int  kb   = lo16 ? 0 : 16;      // B-fragment K chunk base
    const int  col  = jbase + n;          // this lane's hidden/gate column

    v8f acc0 = splat8(bias[col]);
    v8f acc1 = splat8(bias[HID + col]);
    v8f acc2 = splat8(bias[2 * HID + col]);
    v8f acc3 = splat8(bias[3 * HID + col]);

    // ---- segment 1 ----
    {
        const unsigned o0 = (unsigned)(col)           * LDW1 + kb;
        const unsigned o1 = (unsigned)(HID + col)     * LDW1 + kb;
        const unsigned o2 = (unsigned)(2 * HID + col) * LDW1 + kb;
        const unsigned o3 = (unsigned)(3 * HID + col) * LDW1 + kb;
        const __bf16* arow = a1 + n * LD_A1 + ka;
        #pragma unroll 2
        for (int k0 = 0; k0 < K1; k0 += 32) {
            v16bf a = make_frag(ld8(arow + k0), ld8(arow + k0 + 8));
            v16bf b;
            b = make_frag(ld8(w1 + o0 + k0), ld8(w1 + o0 + k0 + 8)); acc0 = WMMA_BF16(a, b, acc0);
            b = make_frag(ld8(w1 + o1 + k0), ld8(w1 + o1 + k0 + 8)); acc1 = WMMA_BF16(a, b, acc1);
            b = make_frag(ld8(w1 + o2 + k0), ld8(w1 + o2 + k0 + 8)); acc2 = WMMA_BF16(a, b, acc2);
            b = make_frag(ld8(w1 + o3 + k0), ld8(w1 + o3 + k0 + 8)); acc3 = WMMA_BF16(a, b, acc3);
        }
    }
    // ---- segment 2 (layer 1 recurrent path) ----
    if constexpr (K2 > 0) {
        const unsigned o0 = (unsigned)(col)           * LDW2 + kb;
        const unsigned o1 = (unsigned)(HID + col)     * LDW2 + kb;
        const unsigned o2 = (unsigned)(2 * HID + col) * LDW2 + kb;
        const unsigned o3 = (unsigned)(3 * HID + col) * LDW2 + kb;
        const __bf16* arow = a2 + n * LD_A2 + ka;
        #pragma unroll 2
        for (int k0 = 0; k0 < K2; k0 += 32) {
            v16bf a = make_frag(ld8(arow + k0), ld8(arow + k0 + 8));
            v16bf b;
            b = make_frag(ld8(w2 + o0 + k0), ld8(w2 + o0 + k0 + 8)); acc0 = WMMA_BF16(a, b, acc0);
            b = make_frag(ld8(w2 + o1 + k0), ld8(w2 + o1 + k0 + 8)); acc1 = WMMA_BF16(a, b, acc1);
            b = make_frag(ld8(w2 + o2 + k0), ld8(w2 + o2 + k0 + 8)); acc2 = WMMA_BF16(a, b, acc2);
            b = make_frag(ld8(w2 + o3 + k0), ld8(w2 + o3 + k0 + 8)); acc3 = WMMA_BF16(a, b, acc3);
        }
    }

    // ---- fused gate nonlinearity + cell update (c lives in LDS) ----
    #pragma unroll
    for (int r = 0; r < 8; ++r) {
        const int m    = r + (lo16 ? 0 : 8);
        const int cidx = m * HID + col;
        const float iv = fast_sigmoid(acc0[r]);
        const float fv = fast_sigmoid(acc1[r]);
        const float gv = fast_tanh(acc2[r]);
        const float ov = fast_sigmoid(acc3[r]);
        const float c  = fv * cst[cidx] + iv * gv;
        cst[cidx] = c;
        hout[m * LDO + col] = (__bf16)(ov * fast_tanh(c));
    }
}

// ---------------------------------------------------------------------------
// Persistent kernel: one workgroup per 16-row batch tile runs both LSTM layers
// across all 512 timesteps. A0 = [x_t | h0] double-buffered; layer-0 writes
// h0_t into A0nxt, which layer-1 reads in place. x_{t+1} staged during layer-0
// phase -> only 2 barriers per timestep. No global synchronization anywhere.
// ---------------------------------------------------------------------------
__global__ __launch_bounds__(NTHR) void lstm_persist_kernel(
    const float*  __restrict__ x,       // [B][T][IN_DIM] fp32
    const __bf16* __restrict__ wbf,     // packed/converted weights (ws)
    const float*  __restrict__ bias0,   // b_ih0 + b_hh0 (ws)
    const float*  __restrict__ bias1,   // b_ih1 + b_hh1 (ws)
    const float*  __restrict__ W1, const float* __restrict__ b1,
    const float*  __restrict__ W2, const float* __restrict__ b2,
    float* __restrict__ out)
{
    __shared__ alignas(16) unsigned char smem[SM_TOTAL];
    __bf16* const a0a = (__bf16*)(smem + SM_A0A);
    __bf16* const a0b = (__bf16*)(smem + SM_A0B);
    __bf16* const h1a = (__bf16*)(smem + SM_H1A);
    __bf16* const h1b = (__bf16*)(smem + SM_H1B);
    float*  const c0  = (float*)(smem + SM_C0);
    float*  const c1  = (float*)(smem + SM_C1);

    const int tid   = threadIdx.x;
    const int wave  = tid >> 5;
    const int bbase = blockIdx.x * BT;

    const __bf16* Wc0  = wbf + OFS_WC0;    // [2048][576] = [W_ih0 | W_hh0]
    const __bf16* Wih1 = wbf + OFS_WIH1;   // [2048][512]
    const __bf16* Whh1 = wbf + OFS_WHH1;   // [2048][512]

    // zero entire LDS state (h regions, c; x regions are rewritten each step)
    for (int i = tid; i < SM_TOTAL / 4; i += NTHR)
        ((unsigned int*)smem)[i] = 0u;

    // uniform double-buffer pointers, swapped with SALU each step
    __bf16* a0cur = a0a;  __bf16* a0nxt = a0b;
    __bf16* h1cur = h1a;  __bf16* h1nxt = h1b;

    __syncthreads();
    // stage x_0 into A0cur cols [0,64)
    for (int i = tid; i < BT * IN_DIM; i += NTHR) {
        const int row = i >> 6, d = i & 63;
        a0cur[row * LDA0 + d] =
            (__bf16)x[(size_t)(bbase + row) * SEQ_T * IN_DIM + d];
    }
    __syncthreads();

    for (int t = 0; t < SEQ_T; ++t) {
        // layer 0: gates = bias0 + [x_t|h0prev](16x576) @ Wcat0^T
        // reads A0cur, writes h0_t into A0nxt cols [64,576)
        lstm_tile_step<KCAT, LDA0, KCAT, 0, 1, 1, LDA0>(
            a0cur, Wc0, nullptr, nullptr, bias0, c0, a0nxt + IN_DIM, wave * 16);
        lstm_tile_step<KCAT, LDA0, KCAT, 0, 1, 1, LDA0>(
            a0cur, Wc0, nullptr, nullptr, bias0, c0, a0nxt + IN_DIM, (wave + NW) * 16);

        // stage x_{t+1} into A0nxt cols [0,64) (disjoint from layer-0 output
        // region and from anything read this step; uniform branch)
        if (t + 1 < SEQ_T) {
            for (int i = tid; i < BT * IN_DIM; i += NTHR) {
                const int row = i >> 6, d = i & 63;
                a0nxt[row * LDA0 + d] =
                    (__bf16)x[((size_t)(bbase + row) * SEQ_T + (t + 1)) * IN_DIM + d];
            }
        }
        __syncthreads();

        // layer 1: gates = bias1 + h0_t(16x512)@Wih1^T + h1prev(16x512)@Whh1^T
        lstm_tile_step<HID, LDA0, HID, HID, LDH1, HID, LDH1>(
            a0nxt + IN_DIM, Wih1, h1cur, Whh1, bias1, c1, h1nxt, wave * 16);
        lstm_tile_step<HID, LDA0, HID, HID, LDH1, HID, LDH1>(
            a0nxt + IN_DIM, Wih1, h1cur, Whh1, bias1, c1, h1nxt, (wave + NW) * 16);
        __syncthreads();

        // swap double buffers (uniform, SALU-only)
        __bf16* t0 = a0cur; a0cur = a0nxt; a0nxt = t0;
        __bf16* t1 = h1cur; h1cur = h1nxt; h1nxt = t1;
    }

    // ---- MLP head (tiny, runs once): hidden = relu(h_last @ W1^T + b1) ----
    const __bf16* hfin = h1cur;
    float* hid = (float*)smem;                 // reuse A0a region (16*260*4 B)
    for (int i = tid; i < BT * 256; i += NTHR) {
        const int row = i >> 8;
        const int u   = i & 255;
        const float*  w  = W1 + (size_t)u * HID;
        const __bf16* hr = hfin + row * LDH1;
        float s = b1[u];
        #pragma unroll 8
        for (int k = 0; k < HID; ++k) s += (float)hr[k] * w[k];
        hid[row * 260 + u] = fmaxf(s, 0.0f);
    }
    __syncthreads();
    // out = sigmoid(hidden @ W2^T + b2)
    if (tid < BT) {
        float s = b2[0];
        #pragma unroll 8
        for (int u = 0; u < 256; ++u) s += hid[tid * 260 + u] * W2[u];
        out[bbase + tid] = fast_sigmoid(s);
    }
}

// ---------------------------------------------------------------------------
// Pre-pass kernels: weight packing/conversion to bf16, bias folding
// ---------------------------------------------------------------------------
__global__ void pack0_kernel(const float* __restrict__ wih,   // [2048][64]
                             const float* __restrict__ whh,   // [2048][512]
                             __bf16* __restrict__ dst) {      // [2048][576]
    for (int i = blockIdx.x * blockDim.x + threadIdx.x; i < GATES * KCAT;
         i += gridDim.x * blockDim.x) {
        const int g = i / KCAT, k = i % KCAT;
        const float v = (k < IN_DIM) ? wih[g * IN_DIM + k]
                                     : whh[g * HID + (k - IN_DIM)];
        dst[i] = (__bf16)v;
    }
}

__global__ void f2bf_kernel(const float* __restrict__ src,
                            __bf16* __restrict__ dst, int n) {
    for (int i = blockIdx.x * blockDim.x + threadIdx.x; i < n;
         i += gridDim.x * blockDim.x)
        dst[i] = (__bf16)src[i];
}

__global__ void bias_kernel(const float* __restrict__ a0, const float* __restrict__ b0,
                            const float* __restrict__ a1, const float* __restrict__ b1,
                            float* __restrict__ o0, float* __restrict__ o1) {
    const int i = blockIdx.x * blockDim.x + threadIdx.x;
    if (i < GATES) {
        o0[i] = a0[i] + b0[i];
        o1[i] = a1[i] + b1[i];
    }
}

// ---------------------------------------------------------------------------
extern "C" void kernel_launch(void* const* d_in, const int* in_sizes, int n_in,
                              void* d_out, int out_size, void* d_ws, size_t ws_size,
                              hipStream_t stream) {
    (void)in_sizes; (void)n_in; (void)out_size; (void)ws_size;

    const float* x    = (const float*)d_in[0];
    const float* Wih0 = (const float*)d_in[1];
    const float* Whh0 = (const float*)d_in[2];
    const float* bih0 = (const float*)d_in[3];
    const float* bhh0 = (const float*)d_in[4];
    const float* Wih1 = (const float*)d_in[5];
    const float* Whh1 = (const float*)d_in[6];
    const float* bih1 = (const float*)d_in[7];
    const float* bhh1 = (const float*)d_in[8];
    const float* W1   = (const float*)d_in[9];
    const float* b1   = (const float*)d_in[10];
    const float* W2   = (const float*)d_in[11];
    const float* b2   = (const float*)d_in[12];
    float* out = (float*)d_out;

    __bf16* wbf   = (__bf16*)d_ws;
    float*  bias0 = (float*)((char*)d_ws + OFS_BIAS0_BYTES);
    float*  bias1 = (float*)((char*)d_ws + OFS_BIAS1_BYTES);

    hipLaunchKernelGGL(pack0_kernel, dim3(2048), dim3(256), 0, stream,
                       Wih0, Whh0, wbf + OFS_WC0);
    hipLaunchKernelGGL(f2bf_kernel, dim3(2048), dim3(256), 0, stream,
                       Wih1, wbf + OFS_WIH1, N_WHH);
    hipLaunchKernelGGL(f2bf_kernel, dim3(2048), dim3(256), 0, stream,
                       Whh1, wbf + OFS_WHH1, N_WHH);
    hipLaunchKernelGGL(bias_kernel, dim3((GATES + 255) / 256), dim3(256), 0, stream,
                       bih0, bhh0, bih1, bhh1, bias0, bias1);

    hipLaunchKernelGGL(lstm_persist_kernel, dim3(BATCH / BT), dim3(NTHR), 0, stream,
                       x, wbf, bias0, bias1, W1, b1, W2, b2, out);
}
